// EncoderLayer_30786325578059
// MI455X (gfx1250) — matrix operand order repaired and checked by reading, not compiled, not run on hardware
//
#include <hip/hip_runtime.h>
#include <hip/hip_bf16.h>

// ---------------------------------------------------------------------------
// Types for CDNA5 WMMA (gfx1250, wave32)
// ---------------------------------------------------------------------------
typedef __bf16 bf16_t;
typedef __attribute__((ext_vector_type(16))) __bf16 v16bf;
typedef __attribute__((ext_vector_type(8)))  float  v8f;

#define LN_EPS 1e-6f

__device__ __forceinline__ float wave_sum32(float x) {
#pragma unroll
  for (int off = 16; off > 0; off >>= 1) x += __shfl_xor(x, off, 32);
  return x;
}

// ---------------------------------------------------------------------------
// Transpose f32 (R x C) -> bf16 (C x R)   (weights into "Bt" N-major layout)
// ---------------------------------------------------------------------------
__global__ __launch_bounds__(256)
void transpose_f32_to_bf16(const float* __restrict__ W, bf16_t* __restrict__ Wt,
                           int R, int C) {
  __shared__ float tile[32][33];
  const int c0 = blockIdx.x * 32, r0 = blockIdx.y * 32;
  const int tx = threadIdx.x & 31, ty = threadIdx.x >> 5;
#pragma unroll
  for (int i = ty; i < 32; i += 8)
    tile[i][tx] = W[(size_t)(r0 + i) * C + c0 + tx];
  __syncthreads();
#pragma unroll
  for (int i = ty; i < 32; i += 8)
    Wt[(size_t)(c0 + i) * R + r0 + tx] = (bf16_t)tile[tx][i];
}

// ---------------------------------------------------------------------------
// Row layernorm over D=1024, emit bf16
// ---------------------------------------------------------------------------
__global__ __launch_bounds__(256)
void layernorm_to_bf16(const float* __restrict__ in, const float* __restrict__ g,
                       const float* __restrict__ bb, bf16_t* __restrict__ out) {
  const int row = blockIdx.x;
  const int tid = threadIdx.x, lane = tid & 31, wv = tid >> 5;
  const float* x = in + (size_t)row * 1024;
  __shared__ float red[8];
  float v[4];
  float s = 0.f;
#pragma unroll
  for (int j = 0; j < 4; ++j) { v[j] = x[tid + 256 * j]; s += v[j]; }
  float ws = wave_sum32(s);
  if (!lane) red[wv] = ws;
  __syncthreads();
  float mu = 0.f;
#pragma unroll
  for (int i = 0; i < 8; ++i) mu += red[i];
  mu *= (1.f / 1024.f);
  __syncthreads();
  float q = 0.f;
#pragma unroll
  for (int j = 0; j < 4; ++j) { float d = v[j] - mu; q += d * d; }
  ws = wave_sum32(q);
  if (!lane) red[wv] = ws;
  __syncthreads();
  float var = 0.f;
#pragma unroll
  for (int i = 0; i < 8; ++i) var += red[i];
  var *= (1.f / 1024.f);
  const float rs = rsqrtf(var + LN_EPS);
#pragma unroll
  for (int j = 0; j < 4; ++j) {
    const int c = tid + 256 * j;
    out[(size_t)row * 1024 + c] = (bf16_t)((v[j] - mu) * rs * g[c] + bb[c]);
  }
}

// ---------------------------------------------------------------------------
// eta pre-pass: eta[row][h] = sigmoid(x_row . lr_w[h] + lr_b[h]) / 32
// ---------------------------------------------------------------------------
__global__ __launch_bounds__(256)
void eta_kernel(const bf16_t* __restrict__ xbf, const float* __restrict__ lr_w,
                const float* __restrict__ lr_b, float* __restrict__ eta) {
  const int row = blockIdx.x;
  const int tid = threadIdx.x, lane = tid & 31, wv = tid >> 5;
  __shared__ float xs[1024];
  for (int i = tid; i < 1024; i += 256) xs[i] = (float)xbf[(size_t)row * 1024 + i];
  __syncthreads();
  for (int h = wv; h < 32; h += 8) {
    float s = 0.f;
    for (int c = lane; c < 1024; c += 32) s += xs[c] * lr_w[h * 1024 + c];
    s = wave_sum32(s);
    if (!lane) {
      const float t = 1.f / (1.f + expf(-(s + lr_b[h])));
      eta[(size_t)row * 32 + h] = t * (1.f / 32.f);  // BASE_LR=1, /HD
    }
  }
}

// ---------------------------------------------------------------------------
// RoPE in place on Q and K (B,L,D) layout; one thread per (b,l,h,i) pair
// ---------------------------------------------------------------------------
__global__ __launch_bounds__(256)
void rope_kernel(float* __restrict__ Q, float* __restrict__ K) {
  const int idx = blockIdx.x * 256 + threadIdx.x;  // < B*L*512
  const int row = idx >> 9;                        // b*2048 + l
  const int t   = idx & 511;
  const int hh  = t >> 4;
  const int i   = t & 15;
  const int l   = row & 2047;
  const size_t base = (size_t)row * 1024 + hh * 32;
  const float f = (float)l * powf(10000.f, -(float)i * (1.f / 16.f));
  float sn, cs;
  sincosf(f, &sn, &cs);
  float q0 = Q[base + i], q1 = Q[base + 16 + i];
  Q[base + i]      = q0 * cs - q1 * sn;
  Q[base + 16 + i] = q1 * cs + q0 * sn;
  float k0 = K[base + i], k1 = K[base + 16 + i];
  K[base + i]      = k0 * cs - k1 * sn;
  K[base + 16 + i] = k1 * cs + k0 * sn;
}

// ---------------------------------------------------------------------------
// bf16 WMMA GEMM: C(MxN) = A(MxK) * Bt(NxK)^T  (+ epilogue)
// 128x128 block tile, BK=32, 8 waves, wave tile 32x64 (2x4 16x16 WMMA tiles)
// EPI: 0 = f32 store; 1 = +addm, f32 store; 2 = +bias, relu, bf16 store;
//      3 = +bias +addm, f32 store
// ---------------------------------------------------------------------------
union FragBF { v16bf v; struct { uint4 lo, hi; } u; };

template <int EPI>
__global__ __launch_bounds__(256)
void gemm_bf16_wmma(const bf16_t* __restrict__ A, const bf16_t* __restrict__ Bt,
                    const float* __restrict__ bias, const float* __restrict__ addm,
                    float* __restrict__ Cf, bf16_t* __restrict__ Cb,
                    int Msz, int Nsz, int Ksz) {
  __shared__ bf16_t sA[128][40];
  __shared__ bf16_t sB[128][40];
  const int tid  = threadIdx.x;
  const int lane = tid & 31;
  const int wave = tid >> 5;
  const int bm = blockIdx.y * 128, bn = blockIdx.x * 128;
  const int wm = (wave >> 1) * 32, wn = (wave & 1) * 64;

  v8f acc[2][4];
#pragma unroll
  for (int i = 0; i < 2; ++i)
#pragma unroll
    for (int j = 0; j < 4; ++j)
#pragma unroll
      for (int t = 0; t < 8; ++t) acc[i][j][t] = 0.f;

  const int rrow = tid >> 1;
  const int coff = (tid & 1) * 16;
  const int r  = lane & 15;
  const int gg = lane >> 4;

  for (int k0 = 0; k0 < Ksz; k0 += 32) {
    // Stage 128x32 bf16 tiles of A and Bt into LDS (32B per thread per tile)
    const uint4* ga = reinterpret_cast<const uint4*>(A + (size_t)(bm + rrow) * Ksz + k0 + coff);
    const uint4* gb = reinterpret_cast<const uint4*>(Bt + (size_t)(bn + rrow) * Ksz + k0 + coff);
    uint4 a0 = ga[0], a1 = ga[1];
    uint4 b0 = gb[0], b1 = gb[1];
    *reinterpret_cast<uint4*>(&sA[rrow][coff])     = a0;
    *reinterpret_cast<uint4*>(&sA[rrow][coff + 8]) = a1;
    *reinterpret_cast<uint4*>(&sB[rrow][coff])     = b0;
    *reinterpret_cast<uint4*>(&sB[rrow][coff + 8]) = b1;
    __syncthreads();

    // A fragment (16x32, 16-bit): lanes 0-15 hold K 0-7 & 16-23, lanes 16-31 K 8-15 & 24-31
    FragBF af[2];
#pragma unroll
    for (int i = 0; i < 2; ++i) {
      const int row = wm + i * 16 + r;
      af[i].u.lo = *reinterpret_cast<const uint4*>(&sA[row][8 * gg]);
      af[i].u.hi = *reinterpret_cast<const uint4*>(&sA[row][8 * gg + 16]);
    }
    // B fragment (32x16): lanes 0-15 hold K 0-7 & 16-23, lanes 16-31 K 8-15 & 24-31
    FragBF bfr[4];
#pragma unroll
    for (int j = 0; j < 4; ++j) {
      const int row = wn + j * 16 + r;
      bfr[j].u.lo = *reinterpret_cast<const uint4*>(&sB[row][8 * gg]);
      bfr[j].u.hi = *reinterpret_cast<const uint4*>(&sB[row][8 * gg + 16]);
    }
#pragma unroll
    for (int i = 0; i < 2; ++i)
#pragma unroll
      for (int j = 0; j < 4; ++j)
        acc[i][j] = __builtin_amdgcn_wmma_f32_16x16x32_bf16(
            false, af[i].v, false, bfr[j].v, (short)0, acc[i][j], false, false);
    __syncthreads();
  }

  // C/D layout: lane n = lane&15; VGPR t -> row t + 8*(lane>>4)
#pragma unroll
  for (int i = 0; i < 2; ++i) {
#pragma unroll
    for (int j = 0; j < 4; ++j) {
      const int col = bn + wn + j * 16 + (lane & 15);
#pragma unroll
      for (int t = 0; t < 8; ++t) {
        const int row = bm + wm + i * 16 + (lane >> 4) * 8 + t;
        const size_t idx = (size_t)row * Nsz + col;
        float v = acc[i][j][t];
        if (EPI == 0) {
          Cf[idx] = v;
        } else if (EPI == 1) {
          Cf[idx] = v + addm[idx];
        } else if (EPI == 2) {
          v += bias[col];
          Cb[idx] = (bf16_t)fmaxf(v, 0.f);
        } else {
          Cf[idx] = v + bias[col] + addm[idx];
        }
      }
    }
  }
}

// ---------------------------------------------------------------------------
// TTT scan: one workgroup per (b,h); 512 threads = 16 waves (wave m = row)
// State W (32x32) and b (32) in LDS; 128 sequential minibatch steps.
// Output written in place over V (already final (B,L,D) layout).
// ---------------------------------------------------------------------------
__global__ __launch_bounds__(512)
void ttt_scan_kernel(const float* __restrict__ Q, const float* __restrict__ K,
                     float* __restrict__ V, const float* __restrict__ eta,
                     const float* __restrict__ W1, const float* __restrict__ b1,
                     const float* __restrict__ lng, const float* __restrict__ lnb,
                     const float* __restrict__ tdelta) {
  const int bh = blockIdx.x;  // 0..255
  const int b  = bh >> 5;
  const int h  = bh & 31;
  const int tid = threadIdx.x;
  const int m = tid >> 5;   // row 0..15
  const int d = tid & 31;   // feature 0..31

  __shared__ float sW[32][33];
  __shared__ float skk[16][33], sq[16][33], sg[16][33];
  __shared__ float sS[16][16];
  __shared__ float sbv[32];
  __shared__ float ee[16], tok[16];

  for (int i = tid; i < 1024; i += 512) sW[i >> 5][i & 31] = W1[h * 1024 + i];
  if (tid < 32) sbv[tid] = b1[h * 32 + tid];
  if (tid < 16) tok[tid] = fmaxf(1.f / (float)(tid + 1) + tdelta[tid], 0.f);
  const float lwv = lng[h * 32 + d];
  const float lbv = lnb[h * 32 + d];
  __syncthreads();
  const float tok15 = tok[15];

  for (int n = 0; n < 128; ++n) {
    const size_t base = ((size_t)(b * 2048 + n * 16 + m)) * 1024 + h * 32 + d;
    const float qv = Q[base], kv = K[base], vv = V[base];
    skk[m][d] = kv;
    sq[m][d]  = qv;
    if (tid < 16) ee[tid] = eta[(size_t)(b * 2048 + n * 16 + tid) * 32 + h];
    __syncthreads();

    // Z1 = k@W + b   and  zq = q@W + b
    float z = sbv[d], zq = sbv[d];
#pragma unroll 8
    for (int c = 0; c < 32; ++c) {
      const float w = sW[c][d];
      z  += skk[m][c] * w;
      zq += sq[m][c] * w;
    }
    // grad = ln_l2_bwd(Z1, v-k)  (rowwise over HD=32 == one wave)
    const float mu  = wave_sum32(z) * (1.f / 32.f);
    const float df  = z - mu;
    const float var = wave_sum32(df * df) * (1.f / 32.f);
    const float std_ = sqrtf(var + LN_EPS);
    const float xh  = df / std_;
    const float tgt = vv - kv;
    const float gxh = (xh * lwv + lbv - tgt) * lwv;
    const float s1 = wave_sum32(gxh);
    const float s2 = wave_sum32(gxh * xh);
    const float g = (32.f * gxh - s1 - xh * s2) / (32.f * std_);
    sg[m][d] = g;
    __syncthreads();

    // S[i][j] = (i>=j) ? tok[i]*ee[j]*(q_i.k_j + 1) : 0
    // (covers both (e*Attn1)@grad and the tril*e@grad term of b_bar)
    if (tid < 256) {
      const int i = tid >> 4, j = tid & 15;
      float sij = 0.f;
      if (i >= j) {
        float dot = 0.f;
#pragma unroll 8
        for (int c = 0; c < 32; ++c) dot += sq[i][c] * skk[j][c];
        sij = tok[i] * ee[j] * (dot + 1.f);
      }
      sS[i][j] = sij;
    }
    __syncthreads();

    // Z1_bar = q@W + b - S@grad ; out = q + ln_fwd(Z1_bar)
    float zb = zq;
#pragma unroll
    for (int j = 0; j < 16; ++j) zb -= sS[m][j] * sg[j][d];
    const float mu2  = wave_sum32(zb) * (1.f / 32.f);
    const float d2   = zb - mu2;
    const float var2 = wave_sum32(d2 * d2) * (1.f / 32.f);
    V[base] = qv + d2 * rsqrtf(var2 + LN_EPS) * lwv + lbv;

    // W -= (last_e*k)^T @ grad ; b -= sum(last_e*grad)
    float a0 = 0.f, a1 = 0.f, bacc = 0.f;
#pragma unroll
    for (int mp = 0; mp < 16; ++mp) {
      const float w = tok15 * ee[mp] * sg[mp][d];
      a0   += w * skk[mp][m];
      a1   += w * skk[mp][m + 16];
      bacc += w;
    }
    sW[m][d]      -= a0;
    sW[m + 16][d] -= a1;
    if (m == 0) sbv[d] -= bacc;
    __syncthreads();
  }
}

// ---------------------------------------------------------------------------
// Launcher
// ---------------------------------------------------------------------------
extern "C" void kernel_launch(void* const* d_in, const int* in_sizes, int n_in,
                              void* d_out, int out_size, void* d_ws, size_t ws_size,
                              hipStream_t stream) {
  (void)in_sizes; (void)n_in; (void)out_size; (void)ws_size;
  const float* enc    = (const float*)d_in[0];
  const float* ln0_g  = (const float*)d_in[1];
  const float* ln0_b  = (const float*)d_in[2];
  const float* Wq     = (const float*)d_in[3];
  const float* Wk     = (const float*)d_in[4];
  const float* Wv     = (const float*)d_in[5];
  const float* Wo     = (const float*)d_in[6];
  const float* W1     = (const float*)d_in[7];
  const float* b1     = (const float*)d_in[8];
  const float* tdelta = (const float*)d_in[9];
  const float* lr_w   = (const float*)d_in[10];
  const float* lr_b   = (const float*)d_in[11];
  const float* ttt_g  = (const float*)d_in[12];
  const float* ttt_b  = (const float*)d_in[13];
  const float* post_g = (const float*)d_in[14];
  const float* post_b = (const float*)d_in[15];
  const float* ffn_g  = (const float*)d_in[16];
  const float* ffn_b  = (const float*)d_in[17];
  const float* fw1    = (const float*)d_in[18];
  const float* fb1    = (const float*)d_in[19];
  const float* fw2    = (const float*)d_in[20];
  const float* fb2    = (const float*)d_in[21];
  float* out = (float*)d_out;

  const int R = 16384;  // B*L
  char* ws = (char*)d_ws;
  const size_t MB = 1ull << 20;
  bf16_t* wqT  = (bf16_t*)(ws + 0 * MB);     // 1024x1024 (N x K) 2MB
  bf16_t* wkT  = (bf16_t*)(ws + 2 * MB);
  bf16_t* wvT  = (bf16_t*)(ws + 4 * MB);
  bf16_t* woT  = (bf16_t*)(ws + 6 * MB);
  bf16_t* fw1T = (bf16_t*)(ws + 8 * MB);     // 2048x1024 4MB
  bf16_t* fw2T = (bf16_t*)(ws + 12 * MB);    // 1024x2048 4MB
  bf16_t* xbf  = (bf16_t*)(ws + 16 * MB);    // 16384x1024 bf16, 32MB
  float*  etab = (float*)(ws + 48 * MB);     // 16384x32 f32, 2MB
  float*  Qf   = (float*)(ws + 50 * MB);     // 64MB
  float*  Kf   = (float*)(ws + 114 * MB);    // 64MB
  float*  Vf   = (float*)(ws + 178 * MB);    // 64MB; reused as scan out, then h3
  float*  r2   = (float*)(ws + 242 * MB);    // 64MB
  bf16_t* h1   = (bf16_t*)(ws + 306 * MB);   // 32MB
  bf16_t* h2   = (bf16_t*)(ws + 338 * MB);   // 32MB
  bf16_t* h3   = (bf16_t*)Vf;                // 16384x2048 bf16, 64MB (scan out dead)

  // 1) Weights -> bf16 transposed (N-major)
  transpose_f32_to_bf16<<<dim3(32, 32), 256, 0, stream>>>(Wq, wqT, 1024, 1024);
  transpose_f32_to_bf16<<<dim3(32, 32), 256, 0, stream>>>(Wk, wkT, 1024, 1024);
  transpose_f32_to_bf16<<<dim3(32, 32), 256, 0, stream>>>(Wv, wvT, 1024, 1024);
  transpose_f32_to_bf16<<<dim3(32, 32), 256, 0, stream>>>(Wo, woT, 1024, 1024);
  transpose_f32_to_bf16<<<dim3(64, 32), 256, 0, stream>>>(fw1, fw1T, 1024, 2048);
  transpose_f32_to_bf16<<<dim3(32, 64), 256, 0, stream>>>(fw2, fw2T, 2048, 1024);

  // 2) LN0 -> bf16 x
  layernorm_to_bf16<<<R, 256, 0, stream>>>(enc, ln0_g, ln0_b, xbf);

  // 3) eta
  eta_kernel<<<R, 256, 0, stream>>>(xbf, lr_w, lr_b, etab);

  // 4) Q/K/V GEMMs + RoPE
  gemm_bf16_wmma<0><<<dim3(8, 128), 256, 0, stream>>>(xbf, wqT, nullptr, nullptr, Qf, nullptr, R, 1024, 1024);
  gemm_bf16_wmma<0><<<dim3(8, 128), 256, 0, stream>>>(xbf, wkT, nullptr, nullptr, Kf, nullptr, R, 1024, 1024);
  gemm_bf16_wmma<0><<<dim3(8, 128), 256, 0, stream>>>(xbf, wvT, nullptr, nullptr, Vf, nullptr, R, 1024, 1024);
  rope_kernel<<<32768, 256, 0, stream>>>(Qf, Kf);

  // 5) TTT scan (writes output in place over Vf)
  ttt_scan_kernel<<<256, 512, 0, stream>>>(Qf, Kf, Vf, etab, W1, b1, ttt_g, ttt_b, tdelta);

  // 6) post-LN, @Wo + residual, ffn-LN, FFN
  layernorm_to_bf16<<<R, 256, 0, stream>>>(Vf, post_g, post_b, h1);
  gemm_bf16_wmma<1><<<dim3(8, 128), 256, 0, stream>>>(h1, woT, nullptr, enc, r2, nullptr, R, 1024, 1024);
  layernorm_to_bf16<<<R, 256, 0, stream>>>(r2, ffn_g, ffn_b, h2);
  gemm_bf16_wmma<2><<<dim3(16, 128), 256, 0, stream>>>(h2, fw1T, fb1, nullptr, nullptr, h3, R, 2048, 1024);
  gemm_bf16_wmma<3><<<dim3(8, 128), 256, 0, stream>>>(h3, fw2T, fb2, r2, out, nullptr, R, 1024, 2048);
}
